// DYConv_2d_61907658604816
// MI455X (gfx1250) — compile-verified, matrix operand order checked by reading, not run on hardware
//
#include <hip/hip_runtime.h>
#include <hip/hip_bf16.h>
#include <math.h>

typedef __bf16 bf16;
typedef __bf16 v16bf __attribute__((ext_vector_type(16)));
typedef __bf16 v8bf  __attribute__((ext_vector_type(8)));
typedef float  v8f   __attribute__((ext_vector_type(8)));

#define EPS 1e-5f
#define TEMP 34.0f

// ---------- helpers ----------

__device__ __forceinline__ int klocal_of(int i, int half) {
    int pair = i >> 1, within = i & 1;
    int kb = (pair < 4) ? (pair * 2) : (16 + (pair - 4) * 2);
    return kb + within + half * 8;
}

__device__ __forceinline__ v8f zero8() {
    v8f z;
#pragma unroll
    for (int j = 0; j < 8; ++j) z[j] = 0.f;
    return z;
}

__device__ __forceinline__ float block_reduce_sum(float v) {
    __shared__ float red[8];
    __syncthreads();
    int lane = threadIdx.x & 31, wv = threadIdx.x >> 5;
#pragma unroll
    for (int m = 16; m >= 1; m >>= 1) v += __shfl_xor(v, m, 32);
    if (lane == 0) red[wv] = v;
    __syncthreads();
    float r = 0.f;
    if (wv == 0) {
        int nw = (int)(blockDim.x >> 5);
        r = (lane < nw) ? red[lane] : 0.f;
#pragma unroll
        for (int m = 4; m >= 1; m >>= 1) r += __shfl_xor(r, m, 32);
    }
    return r; // valid on wave 0
}

// ---------- small kernels ----------

__global__ void k_clear(float* p, int n) {
    int i = blockIdx.x * blockDim.x + threadIdx.x;
    if (i < n) p[i] = 0.f;
}

// BN+ReLU + f32 NCHW output (layer-4 -> triplet tensor)
__global__ void k_prep(const float* __restrict__ in, float* __restrict__ out_f32,
                       const float* __restrict__ scale, const float* __restrict__ shift,
                       int C, int npix, int do_relu) {
    int plane = blockIdx.x;           // b*C + c
    int c = plane % C;
    float sc = scale ? scale[c] : 1.f;
    float sh = shift ? shift[c] : 0.f;
    const float* src = in + (size_t)plane * npix;
    for (int i = threadIdx.x; i < npix; i += blockDim.x) {
        float v = sc * src[i] + sh;
        if (do_relu) v = fmaxf(v, 0.f);
        out_f32[(size_t)plane * npix + i] = v;
    }
}

// BN(optional)+ReLU(optional) + bf16 convert + NCHW -> NHWC(channel-pad 128)
__global__ __launch_bounds__(256) void k_tonhwc(
    const float* __restrict__ in, bf16* __restrict__ out,
    const float* __restrict__ scale, const float* __restrict__ shift,
    int C, int npix, int npixpad, int do_relu) {
    __shared__ bf16 tile[128][66];
    int b = blockIdx.y;
    int pixbase = blockIdx.x * 64;
    int tid = threadIdx.x;
    int px = tid & 63, c0 = tid >> 6;
    int pix = pixbase + px;
    for (int c = c0; c < 128; c += 4) {
        float v = 0.f;
        if (c < C && pix < npix) {
            float sc = scale ? scale[c] : 1.f;
            float sh = shift ? shift[c] : 0.f;
            v = sc * in[((size_t)b * C + c) * npix + pix] + sh;
            if (do_relu) v = fmaxf(v, 0.f);
        }
        tile[c][px] = (bf16)v;
    }
    __syncthreads();
    for (int i = tid; i < 64 * 128; i += 256) {
        int p2 = i >> 7, cc = i & 127;
        out[((size_t)b * npixpad + pixbase + p2) * 128 + cc] = tile[cc][p2];
    }
}

// global average pool from NHWC-padded bf16
__global__ __launch_bounds__(512) void k_pool(const bf16* __restrict__ nhwc,
                                              float* __restrict__ pooled,
                                              int C, int npix, int npixpad) {
    __shared__ float red[512];
    int b = blockIdx.x;
    int c = threadIdx.x & 127, g = threadIdx.x >> 7;
    float s = 0.f;
    for (int p = g; p < npix; p += 4)
        s += (float)nhwc[((size_t)b * npixpad + p) * 128 + c];
    red[threadIdx.x] = s;
    __syncthreads();
    if (threadIdx.x < 128) {
        float tot = red[c] + red[c + 128] + red[c + 256] + red[c + 384];
        if (c < C) pooled[b * C + c] = tot / (float)npix;
    }
}

__global__ void k_x1(const float* __restrict__ pooled, const float* __restrict__ w,
                     const float* __restrict__ bias, float* __restrict__ out) {
    int idx = blockIdx.x * blockDim.x + threadIdx.x;
    if (idx >= 64 * 64) return;
    int b = idx >> 6, o = idx & 63;
    float s = bias[o];
#pragma unroll 4
    for (int c = 0; c < 100; ++c) s += pooled[b * 100 + c] * w[o * 100 + c];
    out[idx] = s;
}

// attention2d: pooled -> relu(fc1) -> fc2 -> softmax(/T)
__global__ void k_attn(const float* __restrict__ pooled, const float* __restrict__ a1w,
                       const float* __restrict__ a2w, const float* __restrict__ a2b,
                       float* __restrict__ attn, int Cin, int hid) {
    __shared__ float sp[128];
    __shared__ float shd[32];
    __shared__ float sl[4];
    int b = blockIdx.x;
    for (int c = threadIdx.x; c < Cin; c += blockDim.x) sp[c] = pooled[b * Cin + c];
    __syncthreads();
    if ((int)threadIdx.x < hid) {
        float s = 0.f;
        for (int c = 0; c < Cin; ++c) s += sp[c] * a1w[threadIdx.x * Cin + c];
        shd[threadIdx.x] = fmaxf(s, 0.f);
    }
    __syncthreads();
    if (threadIdx.x < 4) {
        float s = a2b[threadIdx.x];
        for (int j = 0; j < hid; ++j) s += shd[j] * a2w[threadIdx.x * hid + j];
        sl[threadIdx.x] = s / TEMP;
    }
    __syncthreads();
    if (threadIdx.x == 0) {
        float m = fmaxf(fmaxf(sl[0], sl[1]), fmaxf(sl[2], sl[3]));
        float e0 = expf(sl[0] - m), e1 = expf(sl[1] - m), e2 = expf(sl[2] - m), e3 = expf(sl[3] - m);
        float inv = 1.f / (e0 + e1 + e2 + e3);
        attn[b * 4 + 0] = e0 * inv; attn[b * 4 + 1] = e1 * inv;
        attn[b * 4 + 2] = e2 * inv; attn[b * 4 + 3] = e3 * inv;
    }
}

__global__ void k_aggb(const float* __restrict__ attn, const float* __restrict__ bias,
                       float* __restrict__ aggb, int Cout) {
    int idx = blockIdx.x * blockDim.x + threadIdx.x;
    if (idx >= 64 * 128) return;
    int b = idx >> 7, co = idx & 127;
    float s = 0.f;
    if (co < Cout)
#pragma unroll
        for (int k = 0; k < 4; ++k) s += attn[b * 4 + k] * bias[k * Cout + co];
    aggb[idx] = s;
}

// Aggregate K=4 weight bank per sample, pack into wave32 WMMA A-fragments.
// K ordering: k-tile kt = p*CT + ktc  (p = kh*3+kw, channel tile ktc)
__global__ void k_packA(const float* __restrict__ attn, const float* __restrict__ Wb,
                        bf16* __restrict__ apack, int Cin, int Cout, int Mtiles, int CT) {
    int b = blockIdx.y;
    int mt = blockIdx.x % Mtiles;
    int kt = blockIdx.x / Mtiles;
    int p = kt / CT, ktc = kt - p * CT;
    int lane = threadIdx.x >> 4;   // 0..31
    int i = threadIdx.x & 15;      // 0..15
    int half = lane >> 4;
    int m = mt * 16 + (lane & 15);
    int ci = ktc * 32 + klocal_of(i, half);
    float v = 0.f;
    if (m < Cout && ci < Cin) {
        float a0 = attn[b * 4 + 0], a1 = attn[b * 4 + 1];
        float a2 = attn[b * 4 + 2], a3 = attn[b * 4 + 3];
        size_t wi = ((size_t)m * Cin + ci) * 9 + p;
        size_t st = (size_t)Cout * Cin * 9;
        v = a0 * Wb[wi] + a1 * Wb[wi + st] + a2 * Wb[wi + 2 * st] + a3 * Wb[wi + 3 * st];
    }
    size_t o = (((size_t)b * gridDim.x + blockIdx.x) * 32 + lane) * 16 + i;
    apack[o] = (bf16)v;
}

// ---------- implicit-GEMM conv, bf16 WMMA, NHWC activations ----------
// Block: 1 sample x 128-pixel strip; 8 waves. Double-buffered LDS staging:
// one barrier per K-step; B staging for step k+1 overlaps WMMAs of step k.
template <int CT, int IH, int IW, int OH, int OW, int PAD, int MT, int COUT, int NPIN>
__global__ __launch_bounds__(256) void k_conv(
    const bf16* __restrict__ nhwc, const bf16* __restrict__ apack,
    const bf16* __restrict__ zrow, const float* __restrict__ aggb,
    float* __restrict__ out_raw, float* __restrict__ ssum, float* __restrict__ ssq) {
    constexpr int NPIX = OH * OW;
    constexpr int KT = 9 * CT;

    __shared__ __align__(32) bf16 lb[2][8][32][16];   // double-buffered B fragments

    const int b = blockIdx.y;
    const int strip = blockIdx.x * 128;
    const int tid = threadIdx.x;
    const int lane = tid & 31, wv = tid >> 5;
    const int half = lane >> 4;

    // B-build role: pixel handled by this thread
    const int np = strip + wv * 16 + (lane & 15);
    const int boh = np / OW;
    const int bow = np - boh * OW;
    const bool pv = np < NPIX;

    // compute role
    const int my_mt = wv % MT;
    const int nt0 = (wv / MT) * MT;

    v8f acc[8];
#pragma unroll
    for (int i = 0; i < MT; ++i) acc[i] = zero8();

    const bf16* ap = apack + (((size_t)b * KT) * MT + my_mt) * 512 + (size_t)lane * 16;
    const bf16* xb = nhwc + (size_t)b * NPIN * 128;

    // one shifted source pointer per kernel position (OOB -> zero row)
    const bf16* src[9];
#pragma unroll
    for (int p = 0; p < 9; ++p) {
        const int kh = p / 3, kw = p % 3;
        const int ih = boh + kh - PAD, iw = bow + kw - PAD;
        const bool inb = pv && ih >= 0 && ih < IH && iw >= 0 && iw < IW;
        src[p] = inb ? (xb + (size_t)(ih * IW + iw) * 128 + half * 8) : zrow;
    }

    auto build = [&](int ktl, int buf) {
        const int p = ktl / CT, ct = ktl - (ktl / CT) * CT;   // constants after unroll
        v8bf r0 = *(const v8bf*)(src[p] + ct * 32);
        v8bf r1 = *(const v8bf*)(src[p] + ct * 32 + 16);
        *(v8bf*)(&lb[buf][wv][lane][0]) = r0;
        *(v8bf*)(&lb[buf][wv][lane][8]) = r1;
    };

    build(0, 0);
    __syncthreads();

#pragma unroll
    for (int ktl = 0; ktl < KT; ++ktl) {
        const int cur = ktl & 1;
        if (ktl + 1 < KT) build(ktl + 1, cur ^ 1);

        v16bf af = *(const v16bf*)(ap + (size_t)ktl * MT * 512);
        if (ktl + 2 < KT)
            __builtin_prefetch(ap + (size_t)(ktl + 2) * MT * 512, 0, 0);
#pragma unroll
        for (int nt = 0; nt < MT; ++nt) {
            v16bf bfv = *(const v16bf*)(&lb[cur][nt0 + nt][lane][0]);
            acc[nt] = __builtin_amdgcn_wmma_f32_16x16x32_bf16(
                false, af, false, bfv, (short)0, acc[nt], false, false);
        }
        __syncthreads();
    }

    // epilogue: add per-sample bias, store raw conv, reduce BN stats
#pragma unroll
    for (int r = 0; r < 8; ++r) {
        int co = my_mt * 16 + r + 8 * half;        // C-layout: M = r + 8*(lane>>4)
        bool cok = (co < COUT);
        float bias = cok ? aggb[b * 128 + co] : 0.f;
        float csum = 0.f, csq = 0.f;
#pragma unroll
        for (int nt = 0; nt < MT; ++nt) {
            int pix = strip + (nt0 + nt) * 16 + (lane & 15);
            float v = acc[nt][r] + bias;
            if (cok && pix < NPIX) {
                out_raw[((size_t)b * COUT + co) * NPIX + pix] = v;
                csum += v; csq += v * v;
            }
        }
#pragma unroll
        for (int m = 8; m >= 1; m >>= 1) {
            csum += __shfl_xor(csum, m, 32);
            csq  += __shfl_xor(csq,  m, 32);
        }
        if (cok && (lane & 15) == 0) {
            atomicAdd(&ssum[co], csum);
            atomicAdd(&ssq[co],  csq);
        }
    }
}

__global__ void k_bnfin(const float* __restrict__ ssum, const float* __restrict__ ssq,
                        const float* __restrict__ g, const float* __restrict__ bb,
                        float* __restrict__ scale, float* __restrict__ shift,
                        int C, float cnt) {
    int c = blockIdx.x * blockDim.x + threadIdx.x;
    if (c >= C) return;
    float m = ssum[c] / cnt;
    float v = ssq[c] / cnt - m * m;
    float sc = g[c] * rsqrtf(v + EPS);
    scale[c] = sc;
    shift[c] = bb[c] - m * sc;
}

// ---------- triplet attention ----------

__global__ void k_zpool(const float* __restrict__ t, float* __restrict__ comp,
                        int A1, int A2, int s1, int s2, int sr, int Lr) {
    int idx = blockIdx.x * blockDim.x + threadIdx.x;
    int tot = 64 * A1 * A2;
    if (idx >= tot) return;
    int a2 = idx % A2; int rr = idx / A2; int a1 = rr % A1; int b = rr / A1;
    const float* base = t + (size_t)b * (64 * 44 * 44) + (size_t)a1 * s1 + (size_t)a2 * s2;
    float mx = -3.4e38f, sm = 0.f;
    for (int i = 0; i < Lr; ++i) {
        float v = base[(size_t)i * sr];
        mx = fmaxf(mx, v); sm += v;
    }
    comp[((size_t)(b * 2 + 0) * A1 + a1) * A2 + a2] = mx;
    comp[((size_t)(b * 2 + 1) * A1 + a1) * A2 + a2] = sm / (float)Lr;
}

__global__ void k_conv7(const float* __restrict__ comp, const float* __restrict__ w7,
                        float* __restrict__ sout, float* __restrict__ st,
                        int A1, int A2) {
    int idx = blockIdx.x * blockDim.x + threadIdx.x;
    int tot = 64 * A1 * A2;
    float v = 0.f;
    if (idx < tot) {
        int a2 = idx % A2; int rr = idx / A2; int a1 = rr % A1; int b = rr / A1;
#pragma unroll
        for (int c = 0; c < 2; ++c)
            for (int u = 0; u < 7; ++u) {
                int y = a1 + u - 3; if (y < 0 || y >= A1) continue;
                for (int xk = 0; xk < 7; ++xk) {
                    int z = a2 + xk - 3; if (z < 0 || z >= A2) continue;
                    v += comp[((size_t)(b * 2 + c) * A1 + y) * A2 + z] * w7[(c * 7 + u) * 7 + xk];
                }
            }
        sout[idx] = v;
    }
    float s  = block_reduce_sum(idx < tot ? v : 0.f);
    float s2 = block_reduce_sum(idx < tot ? v * v : 0.f);
    if (threadIdx.x == 0) { atomicAdd(&st[0], s); atomicAdd(&st[1], s2); }
}

__global__ void k_final(const float* __restrict__ t,
                        const float* __restrict__ s_cw, const float* __restrict__ s_hc,
                        const float* __restrict__ s_hw, const float* __restrict__ ts,
                        const float* cw_g, const float* cw_b,
                        const float* hc_g, const float* hc_b,
                        const float* hw_g, const float* hw_b,
                        float* __restrict__ out1) {
    int b = blockIdx.x >> 6, c = blockIdx.x & 63;
    const float c1 = 64.f * 64.f * 44.f, c2 = 64.f * 44.f * 64.f, c3 = 64.f * 44.f * 44.f;
    float m1 = ts[0] / c1, v1 = ts[1] / c1 - m1 * m1;
    float m2 = ts[2] / c2, v2 = ts[3] / c2 - m2 * m2;
    float m3 = ts[4] / c3, v3 = ts[5] / c3 - m3 * m3;
    float sc1 = cw_g[0] * rsqrtf(v1 + EPS), sh1 = cw_b[0] - m1 * sc1;
    float sc2 = hc_g[0] * rsqrtf(v2 + EPS), sh2 = hc_b[0] - m2 * sc2;
    float sc3 = hw_g[0] * rsqrtf(v3 + EPS), sh3 = hw_b[0] - m3 * sc3;
    const float* tp = t + ((size_t)b * 64 + c) * 1936;
    float s = 0.f;
    for (int p = threadIdx.x; p < 1936; p += blockDim.x) {
        int h = p / 44, w = p - h * 44;
        float f1 = 1.f / (1.f + expf(-(sc1 * s_cw[((size_t)b * 64 + c) * 44 + w] + sh1)));
        float f2 = 1.f / (1.f + expf(-(sc2 * s_hc[((size_t)b * 44 + h) * 64 + c] + sh2)));
        float f3 = 1.f / (1.f + expf(-(sc3 * s_hw[((size_t)b * 44 + h) * 44 + w] + sh3)));
        s += tp[p] * (f1 + f2 + f3);
    }
    float tot = block_reduce_sum(s);
    if (threadIdx.x == 0) out1[blockIdx.x] = tot / (3.f * 1936.f);
}

// ---------- host ----------

extern "C" void kernel_launch(void* const* d_in, const int* in_sizes, int n_in,
                              void* d_out, int out_size, void* d_ws, size_t ws_size,
                              hipStream_t stream) {
    const float* x      = (const float*)d_in[0];
    const float* fc3_w  = (const float*)d_in[1];
    const float* fc3_b  = (const float*)d_in[2];
    const float* L[4][7];
    for (int l = 0; l < 4; ++l)
        for (int j = 0; j < 7; ++j) L[l][j] = (const float*)d_in[3 + l * 7 + j];
    const float* cw_w = (const float*)d_in[31];
    const float* cw_g = (const float*)d_in[32];
    const float* cw_b = (const float*)d_in[33];
    const float* hc_w = (const float*)d_in[34];
    const float* hc_g = (const float*)d_in[35];
    const float* hc_b = (const float*)d_in[36];
    const float* hw_w = (const float*)d_in[37];
    const float* hw_g = (const float*)d_in[38];
    const float* hw_b = (const float*)d_in[39];
    float* out = (float*)d_out;

    char* ws = (char*)d_ws;
    size_t off = 0;
    auto alloc = [&](size_t bytes) -> void* {
        void* p = ws + off;
        off = (off + bytes + 255) & ~(size_t)255;
        return p;
    };
    bf16*  nhwcA  = (bf16*) alloc(64ull * 2304 * 128 * 2);
    bf16*  nhwcB  = (bf16*) alloc(64ull * 2304 * 128 * 2);
    float* outraw = (float*)alloc(64ull * 120 * 2304 * 4);
    float* tbuf   = (float*)alloc(64ull * 64 * 44 * 44 * 4);
    bf16*  apack  = (bf16*) alloc(64ull * 36 * 8 * 512 * 2);
    float* pooled = (float*)alloc(64ull * 128 * 4);
    float* attnb  = (float*)alloc(64 * 4 * 4);
    float* aggb   = (float*)alloc(64 * 128 * 4);
    float* stats  = (float*)alloc(256 * 4);
    float* bnsc   = (float*)alloc(128 * 4);
    float* bnsh   = (float*)alloc(128 * 4);
    bf16*  zrow   = (bf16*) alloc(512);
    float* comp   = (float*)alloc(64ull * 2 * 64 * 44 * 4);
    float* s_cw   = (float*)alloc(64ull * 64 * 44 * 4);
    float* s_hc   = (float*)alloc(64ull * 64 * 44 * 4);
    float* s_hw   = (float*)alloc(64ull * 44 * 44 * 4);
    float* tstats = (float*)alloc(8 * 4);

    k_clear<<<1, 128, 0, stream>>>((float*)zrow, 128);

    // ---- input prep: NHWC convert + pooled0; x1 branch ----
    k_tonhwc<<<dim3(36, 64), 256, 0, stream>>>(x, nhwcA, nullptr, nullptr, 100, 2304, 2304, 0);
    k_pool<<<64, 512, 0, stream>>>(nhwcA, pooled, 100, 2304, 2304);
    k_x1<<<16, 256, 0, stream>>>(pooled, fc3_w, fc3_b, out);

    // ---- layer 1: Cin=100 (CT=4) Cout=60 48x48 pad1, MT=4, KT=36 ----
    k_attn<<<64, 64, 0, stream>>>(pooled, L[0][0], L[0][1], L[0][2], attnb, 100, 26);
    k_aggb<<<32, 256, 0, stream>>>(attnb, L[0][4], aggb, 60);
    k_packA<<<dim3(36 * 4, 64), 512, 0, stream>>>(attnb, L[0][3], apack, 100, 60, 4, 4);
    k_clear<<<1, 256, 0, stream>>>(stats, 256);
    k_conv<4, 48, 48, 48, 48, 1, 4, 60, 2304><<<dim3(18, 64), 256, 0, stream>>>(
        nhwcA, apack, zrow, aggb, outraw, stats, stats + 128);
    k_bnfin<<<1, 128, 0, stream>>>(stats, stats + 128, L[0][5], L[0][6], bnsc, bnsh, 60, 64.f * 2304.f);
    k_tonhwc<<<dim3(36, 64), 256, 0, stream>>>(outraw, nhwcB, bnsc, bnsh, 60, 2304, 2304, 1);
    k_pool<<<64, 512, 0, stream>>>(nhwcB, pooled, 60, 2304, 2304);

    // ---- layer 2: Cin=60 (CT=2) Cout=120 48x48 pad1, MT=8, KT=18 ----
    k_attn<<<64, 64, 0, stream>>>(pooled, L[1][0], L[1][1], L[1][2], attnb, 60, 16);
    k_aggb<<<32, 256, 0, stream>>>(attnb, L[1][4], aggb, 120);
    k_packA<<<dim3(18 * 8, 64), 512, 0, stream>>>(attnb, L[1][3], apack, 60, 120, 8, 2);
    k_clear<<<1, 256, 0, stream>>>(stats, 256);
    k_conv<2, 48, 48, 48, 48, 1, 8, 120, 2304><<<dim3(18, 64), 256, 0, stream>>>(
        nhwcB, apack, zrow, aggb, outraw, stats, stats + 128);
    k_bnfin<<<1, 128, 0, stream>>>(stats, stats + 128, L[1][5], L[1][6], bnsc, bnsh, 120, 64.f * 2304.f);
    k_tonhwc<<<dim3(36, 64), 256, 0, stream>>>(outraw, nhwcA, bnsc, bnsh, 120, 2304, 2304, 1);
    k_pool<<<64, 512, 0, stream>>>(nhwcA, pooled, 120, 2304, 2304);

    // ---- layer 3: Cin=120 (CT=4) Cout=120 48->46 pad0, MT=8, KT=36 ----
    k_attn<<<64, 64, 0, stream>>>(pooled, L[2][0], L[2][1], L[2][2], attnb, 120, 31);
    k_aggb<<<32, 256, 0, stream>>>(attnb, L[2][4], aggb, 120);
    k_packA<<<dim3(36 * 8, 64), 512, 0, stream>>>(attnb, L[2][3], apack, 120, 120, 8, 4);
    k_clear<<<1, 256, 0, stream>>>(stats, 256);
    k_conv<4, 48, 48, 46, 46, 0, 8, 120, 2304><<<dim3(17, 64), 256, 0, stream>>>(
        nhwcA, apack, zrow, aggb, outraw, stats, stats + 128);
    k_bnfin<<<1, 128, 0, stream>>>(stats, stats + 128, L[2][5], L[2][6], bnsc, bnsh, 120, 64.f * 2116.f);
    k_tonhwc<<<dim3(34, 64), 256, 0, stream>>>(outraw, nhwcB, bnsc, bnsh, 120, 2116, 2176, 1);
    k_pool<<<64, 512, 0, stream>>>(nhwcB, pooled, 120, 2116, 2176);

    // ---- layer 4: Cin=120 (CT=4) Cout=64 46->44 pad0, MT=4, KT=36 ----
    k_attn<<<64, 64, 0, stream>>>(pooled, L[3][0], L[3][1], L[3][2], attnb, 120, 31);
    k_aggb<<<32, 256, 0, stream>>>(attnb, L[3][4], aggb, 64);
    k_packA<<<dim3(36 * 4, 64), 512, 0, stream>>>(attnb, L[3][3], apack, 120, 64, 4, 4);
    k_clear<<<1, 256, 0, stream>>>(stats, 256);
    k_conv<4, 46, 46, 44, 44, 0, 4, 64, 2176><<<dim3(16, 64), 256, 0, stream>>>(
        nhwcB, apack, zrow, aggb, outraw, stats, stats + 128);
    k_bnfin<<<1, 128, 0, stream>>>(stats, stats + 128, L[3][5], L[3][6], bnsc, bnsh, 64, 64.f * 1936.f);
    k_prep<<<64 * 64, 256, 0, stream>>>(outraw, tbuf, bnsc, bnsh, 64, 1936, 1);

    // ---- triplet attention on tbuf [64,64,44,44] ----
    k_clear<<<1, 64, 0, stream>>>(tstats, 8);
    // cw branch: pool over H -> maps [B, C=64, W=44]
    k_zpool<<<(64 * 64 * 44 + 255) / 256, 256, 0, stream>>>(tbuf, comp, 64, 44, 1936, 1, 44, 44);
    k_conv7<<<(64 * 64 * 44 + 255) / 256, 256, 0, stream>>>(comp, cw_w, s_cw, tstats + 0, 64, 44);
    // hc branch: pool over W -> maps [B, H=44, C=64]
    k_zpool<<<(64 * 44 * 64 + 255) / 256, 256, 0, stream>>>(tbuf, comp, 44, 64, 44, 1936, 1, 44);
    k_conv7<<<(64 * 44 * 64 + 255) / 256, 256, 0, stream>>>(comp, hc_w, s_hc, tstats + 2, 44, 64);
    // hw branch: pool over C -> maps [B, H=44, W=44]
    k_zpool<<<(64 * 44 * 44 + 255) / 256, 256, 0, stream>>>(tbuf, comp, 44, 44, 44, 1, 1936, 64);
    k_conv7<<<(64 * 44 * 44 + 255) / 256, 256, 0, stream>>>(comp, hw_w, s_hw, tstats + 4, 44, 44);

    k_final<<<4096, 256, 0, stream>>>(tbuf, s_cw, s_hc, s_hw, tstats,
                                      cw_g, cw_b, hc_g, hc_b, hw_g, hw_b, out + 4096);

    (void)in_sizes; (void)n_in; (void)out_size; (void)ws_size;
}
